// KI_39170101739842
// MI455X (gfx1250) — matrix operand order repaired and checked by reading, not compiled, not run on hardware
//
#include <hip/hip_runtime.h>
#include <hip/hip_bf16.h>
#include <math.h>

typedef _Float16 v16h __attribute__((ext_vector_type(16)));
typedef float    v8f  __attribute__((ext_vector_type(8)));
typedef float    v2f  __attribute__((ext_vector_type(2)));
typedef int      v4i  __attribute__((vector_size(4*sizeof(int))));

#define AS1 __attribute__((address_space(1)))
#define AS3 __attribute__((address_space(3)))

#define B_  8
#define H_  640
#define W_  368
#define HW  (H_*W_)      // 235520
#define P_  (B_*HW)      // 1884160

__device__ __forceinline__ float lrelu(float x){ return x >= 0.f ? x : 0.01f*x; }

// ---- async global -> LDS 128-bit copy (CDNA5 ASYNCcnt path), hedged ----
#if defined(__has_builtin) && __has_builtin(__builtin_amdgcn_global_load_async_to_lds_b128)
__device__ __forceinline__ void async_cp16(const void* g, void* l){
  __builtin_amdgcn_global_load_async_to_lds_b128(
      (AS1 v4i*)g, (AS3 v4i*)l, 0, 0);
}
#else
__device__ __forceinline__ void async_cp16(const void* g, void* l){
  unsigned loff = (unsigned)(unsigned long)l;   // low 32 bits of flat LDS ptr = LDS offset
  asm volatile("global_load_async_to_lds_b128 %0, %1, off" :: "v"(loff), "v"(g) : "memory");
}
#endif
__device__ __forceinline__ void wait_async0(){
#if defined(__has_builtin) && __has_builtin(__builtin_amdgcn_s_wait_asynccnt)
  __builtin_amdgcn_s_wait_asynccnt(0);
#else
  asm volatile("s_wait_asynccnt 0x0" ::: "memory");
#endif
}

// ---------------------------------------------------------------------------
// Centered ortho inverse-DFT matrix: M[a][b] = exp(+2*pi*i*((a+N/2)%N)*((b+N/2)%N)/N)/sqrt(N)
// Folds ifftshift (input) and fftshift (output) into the operator.
// ---------------------------------------------------------------------------
__global__ void gen_dft(float* __restrict__ mre, float* __restrict__ mim, int N){
  int idx = blockIdx.x*blockDim.x + threadIdx.x;
  if (idx >= N*N) return;
  int a = idx / N, b = idx % N, c = N/2;
  int aa = a + c; if (aa >= N) aa -= N;
  int bb = b + c; if (bb >= N) bb -= N;
  int t = (aa * bb) % N;
  float th = 6.28318530717958647692f * (float)t / (float)N;
  float s  = rsqrtf((float)N);
  mre[idx] = cosf(th) * s;
  mim[idx] = sinf(th) * s;
}

// ---------------------------------------------------------------------------
// Fused k-space front end: 3x3 conv (2->32) + leaky, then 1x1 (32->2) + leaky.
// Implicit GEMM via two v_wmma_f32_16x16x32_f16 per 16-pixel tile.
// ---------------------------------------------------------------------------
__global__ __launch_bounds__(128) void conv_front(
    const float* __restrict__ ksp, const float* __restrict__ wk,
    const float* __restrict__ bk,  const float* __restrict__ w1,
    const float* __restrict__ b1,  float* __restrict__ outre,
    float* __restrict__ outim)
{
  __shared__ float fbuf[4][16][33];          // [wave][pixel][channel(+pad)]
  int wave = threadIdx.x >> 5;
  int lane = threadIdx.x & 31;
  int g = lane >> 4, q = lane & 15;
  int t   = blockIdx.x*4 + wave;             // tile id, 117760 total
  int row = t / 23;                          // flattened (b,h)
  int w0  = (t % 23) * 16;
  int b   = row / H_, h = row % H_;
  int wp  = w0 + q;

  v16h a = {};
  #pragma unroll
  for (int e = 0; e < 16; ++e){
    int k = (e < 8) ? (e + 8*g) : (e - 8 + 16 + 8*g);
    if (k < 18){
      int ch = k / 9, tap = k % 9;
      int hh = h + tap/3 - 1, ww = wp + tap%3 - 1;
      bool ok = (hh >= 0) & (hh < H_) & (ww >= 0) & (ww < W_);
      int hc = min(max(hh,0),H_-1), wc = min(max(ww,0),W_-1);
      float v = ksp[((b*2 + ch)*H_ + hc)*W_ + wc];
      a[e] = (_Float16)(ok ? v : 0.f);
    }
  }
  v16h wb0 = {}, wb1 = {};
  #pragma unroll
  for (int e = 0; e < 16; ++e){
    int k = (e < 8) ? (e + 8*g) : (e - 8 + 16 + 8*g);
    if (k < 18){
      int ch = k/9, tap = k%9;
      wb0[e] = (_Float16)wk[((q     )*2 + ch)*9 + tap];
      wb1[e] = (_Float16)wk[((q + 16)*2 + ch)*9 + tap];
    }
  }
  v8f c0 = {}, c1 = {};
  c0 = __builtin_amdgcn_wmma_f32_16x16x32_f16(false, a, false, wb0, (short)0, c0, false, false);
  c1 = __builtin_amdgcn_wmma_f32_16x16x32_f16(false, a, false, wb1, (short)0, c1, false, false);

  #pragma unroll
  for (int r = 0; r < 8; ++r){
    int m = r + 8*g;
    fbuf[wave][m][q]      = lrelu(c0[r] + bk[q]);
    fbuf[wave][m][q + 16] = lrelu(c1[r] + bk[q + 16]);
  }
  __syncthreads();

  if (lane < 16){
    float sre = b1[0], sim = b1[1];
    #pragma unroll
    for (int ch = 0; ch < 32; ++ch){
      float v = fbuf[wave][lane][ch];
      sre += v * w1[ch];
      sim += v * w1[32 + ch];
    }
    int p = row*W_ + w0 + lane;
    outre[p] = lrelu(sre);
    outim[p] = lrelu(sim);
  }
}

// ---------------------------------------------------------------------------
// Complex DFT-as-GEMM pass with async global->LDS tile staging.
// out[rho][n] = sum_k in[rho][k] * M[k][n];  re = rr - ii, im = ri + ir.
// One 16x16 complex tile per wave; K consumed in chunks of 16.
// elem(rho,k) = (rho/inner)*ostride + (rho%inner)*istride + k*kstride
// aswap=0: global contiguous along k  -> LDS A tile stored [m][k]
// aswap=1: global contiguous along rho-> LDS A tile stored [k][m]
// ---------------------------------------------------------------------------
__global__ __launch_bounds__(256) void fft_gemm(
    const float* __restrict__ inre, const float* __restrict__ inim,
    float* __restrict__ outre, float* __restrict__ outim,
    const float* __restrict__ mre, const float* __restrict__ mim,
    int S, int inner, int ostride, int istride, int kstride,
    int tiles_n, int aswap)
{
  __shared__ __align__(16) float smem[8][4][256];   // [wave][Are,Aim,Bre,Bim][16x16]
  int wave = threadIdx.x >> 5;
  int lane = threadIdx.x & 31;
  int g = lane >> 4, q = lane & 15;
  int t  = blockIdx.x*8 + wave;
  int tm = t / tiles_n, tn = t % tiles_n;
  int ncol = tn*16 + q;

  float* Are = &smem[wave][0][0];
  float* Aim = &smem[wave][1][0];
  float* Bre = &smem[wave][2][0];
  float* Bim = &smem[wave][3][0];

  v8f crr = {}, cii = {}, cri = {}, cir = {};
  for (int k0 = 0; k0 < S; k0 += 16){
    // ---- stage A and B 16x16 tiles into LDS: 2 x b128 per plane ----
    #pragma unroll
    for (int half = 0; half < 2; ++half){
      int j  = half*32 + lane;       // 0..63 transfer slots (16B each)
      int t0 = j >> 2;               // 0..15
      int e4 = (j & 3) * 4;          // 0,4,8,12
      long ga; int la;
      if (aswap == 0){               // rows m=t0, contiguous k
        int rho = tm*16 + t0;
        long bm = (long)(rho / inner) * ostride + (long)(rho % inner) * istride;
        ga = bm + (long)(k0 + e4) * kstride;   // kstride == 1 here
        la = t0*16 + e4;
      } else {                       // k=t0, contiguous rows (istride == 1)
        int rho = tm*16 + e4;
        long bm = (long)(rho / inner) * ostride + (long)(rho % inner) * istride;
        ga = bm + (long)(k0 + t0) * kstride;
        la = t0*16 + e4;
      }
      async_cp16(inre + ga, Are + la);
      async_cp16(inim + ga, Aim + la);
      int gb = (k0 + t0)*S + tn*16 + e4;       // B always contiguous along n
      async_cp16(mre + gb, Bre + (t0*16 + e4));
      async_cp16(mim + gb, Bim + (t0*16 + e4));
    }
    wait_async0();

    // ---- 4 K-substeps of 4: v_wmma_f32_16x16x4_f32 x4 (complex) ----
    #pragma unroll
    for (int kk = 0; kk < 16; kk += 4){
      int ka0 = kk + 2*g, ka1 = ka0 + 1;       // f32 A 16x4 lane layout
      int ia0 = aswap ? (ka0*16 + q) : (q*16 + ka0);
      int ia1 = aswap ? (ka1*16 + q) : (q*16 + ka1);
      v2f are, aim, bre, bim;
      are.x = Are[ia0]; are.y = Are[ia1];
      aim.x = Aim[ia0]; aim.y = Aim[ia1];
      bre.x = Bre[ka0*16 + q]; bre.y = Bre[ka1*16 + q];
      bim.x = Bim[ka0*16 + q]; bim.y = Bim[ka1*16 + q];
      crr = __builtin_amdgcn_wmma_f32_16x16x4_f32(false, are, false, bre, (short)0, crr, false, false);
      cii = __builtin_amdgcn_wmma_f32_16x16x4_f32(false, aim, false, bim, (short)0, cii, false, false);
      cri = __builtin_amdgcn_wmma_f32_16x16x4_f32(false, are, false, bim, (short)0, cri, false, false);
      cir = __builtin_amdgcn_wmma_f32_16x16x4_f32(false, aim, false, bre, (short)0, cir, false, false);
    }
  }
  #pragma unroll
  for (int r = 0; r < 8; ++r){
    int ro = tm*16 + r + 8*g;
    long ob = (long)(ro / inner) * ostride + (long)(ro % inner) * istride;
    outre[ob + (long)ncol*kstride] = crr[r] - cii[r];
    outim[ob + (long)ncol*kstride] = cri[r] + cir[r];
  }
}

// ---------------------------------------------------------------------------
// Batch-axis 8-point centered IDFT + |.| + fp16 1x1 conv2 (2->1) + leaky.
// ---------------------------------------------------------------------------
__global__ void pass5_abs_conv2(
    const float* __restrict__ inre, const float* __restrict__ inim,
    const float* __restrict__ img,  const float* __restrict__ w2,
    const float* __restrict__ b2,   float* __restrict__ outc2)
{
  int hw = blockIdx.x*blockDim.x + threadIdx.x;
  if (hw >= HW) return;
  const float ct[8] = {1.f, 0.70710678f, 0.f, -0.70710678f, -1.f, -0.70710678f, 0.f, 0.70710678f};
  const float st[8] = {0.f, 0.70710678f, 1.f, 0.70710678f, 0.f, -0.70710678f, -1.f, -0.70710678f};
  float xr[8], xi[8];
  #pragma unroll
  for (int bb = 0; bb < 8; ++bb){ xr[bb] = inre[bb*HW + hw]; xi[bb] = inim[bb*HW + hw]; }
  _Float16 w20 = (_Float16)w2[0], w21 = (_Float16)w2[1], b20 = (_Float16)b2[0];
  _Float16 slope = (_Float16)0.01f;
  #pragma unroll
  for (int aa = 0; aa < 8; ++aa){
    float ar = 0.f, ai = 0.f;
    int a4 = (aa + 4) & 7;
    #pragma unroll
    for (int bb = 0; bb < 8; ++bb){
      int t8 = (a4 * ((bb + 4) & 7)) & 7;
      ar += xr[bb]*ct[t8] - xi[bb]*st[t8];
      ai += xr[bb]*st[t8] + xi[bb]*ct[t8];
    }
    ar *= 0.35355339059327376f; ai *= 0.35355339059327376f;   // 1/sqrt(8)
    float mag = sqrtf(ar*ar + ai*ai);
    _Float16 o = w20*(_Float16)mag + w21*(_Float16)img[aa*HW + hw] + b20;
    o = (o >= (_Float16)0.f) ? o : o * slope;                 // leaky in fp16
    outc2[aa*HW + hw] = (float)o;
  }
}

// ---------------------------------------------------------------------------
// Final image-domain 3x3 conv (1->1, pad 1) + leaky.
// ---------------------------------------------------------------------------
__global__ void conv_tail(const float* __restrict__ c2, const float* __restrict__ wi,
                          const float* __restrict__ bi, float* __restrict__ out)
{
  int idx = blockIdx.x*blockDim.x + threadIdx.x;
  if (idx >= P_) return;
  int b = idx / HW, hw = idx % HW;
  int h = hw / W_, w = hw % W_;
  float acc = bi[0];
  #pragma unroll
  for (int tap = 0; tap < 9; ++tap){
    int hh = h + tap/3 - 1, ww = w + tap%3 - 1;
    if (hh >= 0 && hh < H_ && ww >= 0 && ww < W_)
      acc += wi[tap] * c2[b*HW + hh*W_ + ww];
  }
  out[idx] = lrelu(acc);
}

// ---------------------------------------------------------------------------
extern "C" void kernel_launch(void* const* d_in, const int* in_sizes, int n_in,
                              void* d_out, int out_size, void* d_ws, size_t ws_size,
                              hipStream_t stream){
  (void)in_sizes; (void)n_in; (void)out_size; (void)ws_size;
  const float* img = (const float*)d_in[0];
  const float* ksp = (const float*)d_in[1];
  const float* wk  = (const float*)d_in[2];
  const float* bk  = (const float*)d_in[3];
  const float* w1  = (const float*)d_in[4];
  const float* b1  = (const float*)d_in[5];
  const float* w2  = (const float*)d_in[6];
  const float* b2  = (const float*)d_in[7];
  const float* wi  = (const float*)d_in[8];
  const float* bi  = (const float*)d_in[9];

  float* ws = (float*)d_ws;
  float* Rre = ws;                    float* Rim = ws +   (size_t)P_;
  float* Sre = ws + 2*(size_t)P_;     float* Sim = ws + 3*(size_t)P_;
  float* d640re = ws + 4*(size_t)P_;
  float* d640im = d640re + 640*640;
  float* d368re = d640im + 640*640;
  float* d368im = d368re + 368*368;

  gen_dft<<<(640*640+255)/256,256,0,stream>>>(d640re,d640im,640);
  gen_dft<<<(368*368+255)/256,256,0,stream>>>(d368re,d368im,368);

  // k-space convs -> complex planes
  conv_front<<<29440,128,0,stream>>>(ksp,wk,bk,w1,b1,Rre,Rim);

  // IFFT along W: rows = B*H = 5120, K=N=368, contiguous along k (aswap=0)
  fft_gemm<<<920,256,0,stream>>>(Rre,Rim,Sre,Sim,d368re,d368im,
                                 368, 5120, 0, W_, 1, 23, 0);
  // IFFT along H: rows = B*W = 2944, K=N=640, contiguous along rho (aswap=1)
  fft_gemm<<<920,256,0,stream>>>(Sre,Sim,Rre,Rim,d640re,d640im,
                                 640, 368, HW, 1, W_, 40, 1);

  // IFFT along B + abs + fp16 conv2 + leaky (writes into Sre region)
  pass5_abs_conv2<<<(HW+255)/256,256,0,stream>>>(Rre,Rim,img,w2,b2,Sre);

  // final 3x3 conv + leaky
  conv_tail<<<(P_+255)/256,256,0,stream>>>(Sre,wi,bi,(float*)d_out);
}
